// Attention_85109071937879
// MI455X (gfx1250) — matrix operand order repaired and checked by reading, not compiled, not run on hardware
//
#include <hip/hip_runtime.h>

// ---------------------------------------------------------------------------
// GQA attention layer for gfx1250 (MI455X), bf16 WMMA everywhere.
// Round 2: 32x64 GEMM tiles + register double-buffering so WMMAs overlap
// with the next step's global loads (avoid s_wait_loadcnt 0x0 stalls).
// ---------------------------------------------------------------------------

typedef __bf16 bf16_t;
typedef __attribute__((ext_vector_type(16))) __bf16 v16bf;
typedef __attribute__((ext_vector_type(8)))  __bf16 v8bf;
typedef __attribute__((ext_vector_type(8)))  float  v8f;

#define DEVFN static __device__ __forceinline__

constexpr int D_MODEL   = 2048;
constexpr int NUM_HEADS = 32;
constexpr int NUM_KV    = 8;
constexpr int HEAD_DIM  = 64;
constexpr int BATCH     = 2;
constexpr int SEQ       = 2048;
constexpr int MTOT      = BATCH * SEQ;        // 4096 rows for the big GEMMs
constexpr int KVD       = NUM_KV * HEAD_DIM;  // 512

// fp32 -> bf16 round-to-nearest-even.
DEVFN bf16_t f2bf(float f) {
  unsigned u = __builtin_bit_cast(unsigned, f);
  unsigned r = u + 0x7fffu + ((u >> 16) & 1u);
  unsigned short s = (unsigned short)(r >> 16);
  return __builtin_bit_cast(bf16_t, s);
}

// A-fragment (16x32 bf16): lane L holds row M=L%16; elems 0..15 are
// K = 16*(L/16)+j.
DEVFN v16bf load_a_frag(const bf16_t* __restrict__ base, int ld, int lane) {
  const int row = lane & 15, g = lane >> 4;
  const bf16_t* p = base + (size_t)row * ld + 16 * g;
  v8bf lo = *(const v8bf*)p;
  v8bf hi = *(const v8bf*)(p + 8);
  return __builtin_shufflevector(lo, hi, 0,1,2,3,4,5,6,7,8,9,10,11,12,13,14,15);
}

// B-fragment (32x16 bf16) of B = W^T (W row-major [N,K]): lane L holds
// column N=L%16, K = 16*(L/16)+j.
DEVFN v16bf load_b_frag(const bf16_t* __restrict__ base, int ld, int lane) {
  const int n = lane & 15, g = lane >> 4;
  return *(const v16bf*)(base + (size_t)n * ld + 16 * g);
}

DEVFN v8f wmma_bf16(v16bf a, v16bf b, v8f c) {
  return __builtin_amdgcn_wmma_f32_16x16x32_bf16(false, a, false, b,
                                                 (short)0, c, false, false);
}

// ---------------------------------------------------------------------------
// fp32 -> bf16 conversion
// ---------------------------------------------------------------------------
__global__ void __launch_bounds__(256) cvt_kernel(const float* __restrict__ in,
                                                  bf16_t* __restrict__ out, int n) {
  int i = blockIdx.x * blockDim.x + threadIdx.x;
  int stride = gridDim.x * blockDim.x;
  for (; i < n; i += stride) out[i] = f2bf(in[i]);
}

// ---------------------------------------------------------------------------
// GEMM core: one wave computes C[m0:m0+32, n0:n0+64] = A[M,K] @ W[N,K]^T.
// Fragments for step k+32 are loaded before the 8 WMMAs of step k issue,
// so the matrix pipe overlaps the global loads.
// ---------------------------------------------------------------------------
DEVFN void gemm_32x64(const bf16_t* __restrict__ A, const bf16_t* __restrict__ W,
                      int K, int m0, int n0, int lane, v8f acc[8]) {
  const v8f z = {0.f, 0.f, 0.f, 0.f, 0.f, 0.f, 0.f, 0.f};
#pragma unroll
  for (int t = 0; t < 8; ++t) acc[t] = z;

  const bf16_t* a0p = A + (size_t)m0 * K;
  const bf16_t* a1p = A + (size_t)(m0 + 16) * K;
  const bf16_t* wp  = W + (size_t)n0 * K;

  v16bf a0 = load_a_frag(a0p, K, lane);
  v16bf a1 = load_a_frag(a1p, K, lane);
  v16bf b0 = load_b_frag(wp, K, lane);
  v16bf b1 = load_b_frag(wp + (size_t)16 * K, K, lane);
  v16bf b2 = load_b_frag(wp + (size_t)32 * K, K, lane);
  v16bf b3 = load_b_frag(wp + (size_t)48 * K, K, lane);

  for (int k0 = 32; k0 < K; k0 += 32) {
    // issue next step's loads first
    v16bf na0 = load_a_frag(a0p + k0, K, lane);
    v16bf na1 = load_a_frag(a1p + k0, K, lane);
    v16bf nb0 = load_b_frag(wp + k0, K, lane);
    v16bf nb1 = load_b_frag(wp + (size_t)16 * K + k0, K, lane);
    v16bf nb2 = load_b_frag(wp + (size_t)32 * K + k0, K, lane);
    v16bf nb3 = load_b_frag(wp + (size_t)48 * K + k0, K, lane);
    // consume current fragments
    acc[0] = wmma_bf16(a0, b0, acc[0]);
    acc[1] = wmma_bf16(a0, b1, acc[1]);
    acc[2] = wmma_bf16(a0, b2, acc[2]);
    acc[3] = wmma_bf16(a0, b3, acc[3]);
    acc[4] = wmma_bf16(a1, b0, acc[4]);
    acc[5] = wmma_bf16(a1, b1, acc[5]);
    acc[6] = wmma_bf16(a1, b2, acc[6]);
    acc[7] = wmma_bf16(a1, b3, acc[7]);
    a0 = na0; a1 = na1; b0 = nb0; b1 = nb1; b2 = nb2; b3 = nb3;
  }
  acc[0] = wmma_bf16(a0, b0, acc[0]);
  acc[1] = wmma_bf16(a0, b1, acc[1]);
  acc[2] = wmma_bf16(a0, b2, acc[2]);
  acc[3] = wmma_bf16(a0, b3, acc[3]);
  acc[4] = wmma_bf16(a1, b0, acc[4]);
  acc[5] = wmma_bf16(a1, b1, acc[5]);
  acc[6] = wmma_bf16(a1, b2, acc[6]);
  acc[7] = wmma_bf16(a1, b3, acc[7]);
}

// Q projection: q = x @ Wq^T, scaled by 1/sqrt(Dh), stored [B,H,L,64] bf16
__global__ void __launch_bounds__(128) gemm_q_kernel(const bf16_t* __restrict__ A,
                                                     const bf16_t* __restrict__ W,
                                                     bf16_t* __restrict__ Q) {
  const int lane = threadIdx.x & 31;
  const int wave = blockIdx.x * (blockDim.x >> 5) + (threadIdx.x >> 5);
  const int ntn = D_MODEL / 64;
  const int m0 = (wave / ntn) * 32;
  const int n0 = (wave % ntn) * 64;
  v8f acc[8];
  gemm_32x64(A, W, D_MODEL, m0, n0, lane, acc);
  const int cb = n0 + (lane & 15);
#pragma unroll
  for (int mt = 0; mt < 2; ++mt) {
    const int rb = m0 + 16 * mt + 8 * (lane >> 4);
#pragma unroll
    for (int t = 0; t < 4; ++t) {
      const int c = cb + 16 * t, h = c >> 6, d = c & 63;
#pragma unroll
      for (int r = 0; r < 8; ++r) {
        const int row = rb + r, bb = row >> 11, l = row & (SEQ - 1);
        Q[((size_t)(bb * NUM_HEADS + h) * SEQ + l) * HEAD_DIM + d] =
            f2bf(acc[4 * mt + t][r] * 0.125f);  // 1/sqrt(64)
      }
    }
  }
}

// K projection: stored [B,Hkv,L,64] bf16
__global__ void __launch_bounds__(128) gemm_k_kernel(const bf16_t* __restrict__ A,
                                                     const bf16_t* __restrict__ W,
                                                     bf16_t* __restrict__ Ko) {
  const int lane = threadIdx.x & 31;
  const int wave = blockIdx.x * (blockDim.x >> 5) + (threadIdx.x >> 5);
  const int ntn = KVD / 64;
  const int m0 = (wave / ntn) * 32;
  const int n0 = (wave % ntn) * 64;
  v8f acc[8];
  gemm_32x64(A, W, D_MODEL, m0, n0, lane, acc);
  const int cb = n0 + (lane & 15);
#pragma unroll
  for (int mt = 0; mt < 2; ++mt) {
    const int rb = m0 + 16 * mt + 8 * (lane >> 4);
#pragma unroll
    for (int t = 0; t < 4; ++t) {
      const int c = cb + 16 * t, h = c >> 6, d = c & 63;
#pragma unroll
      for (int r = 0; r < 8; ++r) {
        const int row = rb + r, bb = row >> 11, l = row & (SEQ - 1);
        Ko[((size_t)(bb * NUM_KV + h) * SEQ + l) * HEAD_DIM + d] =
            f2bf(acc[4 * mt + t][r]);
      }
    }
  }
}

// V projection: stored TRANSPOSED [B,Hkv,64,L] bf16 (Vt[d][s]) so the PV
// matmul A-fragments are contiguous.
__global__ void __launch_bounds__(128) gemm_v_kernel(const bf16_t* __restrict__ A,
                                                     const bf16_t* __restrict__ W,
                                                     bf16_t* __restrict__ Vt) {
  const int lane = threadIdx.x & 31;
  const int wave = blockIdx.x * (blockDim.x >> 5) + (threadIdx.x >> 5);
  const int ntn = KVD / 64;
  const int m0 = (wave / ntn) * 32;
  const int n0 = (wave % ntn) * 64;
  v8f acc[8];
  gemm_32x64(A, W, D_MODEL, m0, n0, lane, acc);
  const int cb = n0 + (lane & 15);
#pragma unroll
  for (int mt = 0; mt < 2; ++mt) {
    const int rb = m0 + 16 * mt + 8 * (lane >> 4);
#pragma unroll
    for (int t = 0; t < 4; ++t) {
      const int c = cb + 16 * t, h = c >> 6, d = c & 63;
#pragma unroll
      for (int r = 0; r < 8; ++r) {
        const int row = rb + r, bb = row >> 11, l = row & (SEQ - 1);
        Vt[((size_t)(bb * NUM_KV + h) * HEAD_DIM + d) * SEQ + l] =
            f2bf(acc[4 * mt + t][r]);
      }
    }
  }
}

// ---------------------------------------------------------------------------
// Causal flash attention. One wave per (b, h, 16-query block).
// S^T = K @ Q^T (softmax stats per-lane), O^T = V^T @ P^T (P^T via shfl_xor).
// K A-frags are double-buffered one key-block ahead; V A-frag loads issue
// before the softmax VALU chain so its latency hides them.
// ---------------------------------------------------------------------------
__global__ void __launch_bounds__(128) attn_kernel(const bf16_t* __restrict__ Qb,
                                                   const bf16_t* __restrict__ Kb,
                                                   const bf16_t* __restrict__ Vtb,
                                                   bf16_t* __restrict__ attnb) {
  const int lane = threadIdx.x & 31;
  const int wave = blockIdx.x * (blockDim.x >> 5) + (threadIdx.x >> 5);
  const int nqb = SEQ / 16;                       // 128
  const int qb = wave % nqb;
  const int h  = (wave / nqb) % NUM_HEADS;
  const int b  = wave / (nqb * NUM_HEADS);
  const int hk = h >> 2;                          // 4 Q heads per KV head
  const int q0 = qb * 16;

  const bf16_t* qh = Qb  + (size_t)(b * NUM_HEADS + h) * SEQ * HEAD_DIM;
  const bf16_t* kh = Kb  + (size_t)(b * NUM_KV + hk) * SEQ * HEAD_DIM;
  const bf16_t* vh = Vtb + (size_t)(b * NUM_KV + hk) * HEAD_DIM * SEQ;

  // Q^T B-fragments (K dim = head dim, split 0..31 / 32..63)
  const v16bf qf0 = load_b_frag(qh + (size_t)q0 * HEAD_DIM + 0,  HEAD_DIM, lane);
  const v16bf qf1 = load_b_frag(qh + (size_t)q0 * HEAD_DIM + 32, HEAD_DIM, lane);

  const v8f z = {0.f, 0.f, 0.f, 0.f, 0.f, 0.f, 0.f, 0.f};
  v8f o[4] = {z, z, z, z};            // O^T tiles: dim blocks 0..3
  float m_run = -__builtin_inff();
  float l_run = 0.f;
  const int qg = q0 + (lane & 15);    // this lane's (global) query index
  const int sslot = 8 * (lane >> 4);  // key-slot offset within a score tile
  const bool lo16 = (lane & 16) == 0;

  const int nkb = (q0 + 16 + 31) >> 5;

  // current K A-fragments (keys s0..s0+15 x k{0,32}, keys s0+16..31 x k{0,32})
  v16bf kc0 = load_a_frag(kh + 0,                        HEAD_DIM, lane);
  v16bf kc1 = load_a_frag(kh + 32,                       HEAD_DIM, lane);
  v16bf kc2 = load_a_frag(kh + (size_t)16 * HEAD_DIM,      HEAD_DIM, lane);
  v16bf kc3 = load_a_frag(kh + (size_t)16 * HEAD_DIM + 32, HEAD_DIM, lane);

  for (int kb = 0; kb < nkb; ++kb) {
    const int s0 = kb * 32;

    // double-buffer: issue next block's K loads before using current frags
    v16bf kn0 = kc0, kn1 = kc1, kn2 = kc2, kn3 = kc3;
    if (kb + 1 < nkb) {
      const size_t sn = (size_t)(s0 + 32) * HEAD_DIM;
      kn0 = load_a_frag(kh + sn, HEAD_DIM, lane);
      kn1 = load_a_frag(kh + sn + 32, HEAD_DIM, lane);
      kn2 = load_a_frag(kh + sn + (size_t)16 * HEAD_DIM, HEAD_DIM, lane);
      kn3 = load_a_frag(kh + sn + (size_t)16 * HEAD_DIM + 32, HEAD_DIM, lane);
    }

    // S^T tiles: st0 = keys s0..s0+15, st1 = keys s0+16..s0+31
    v8f st0 = z, st1 = z;
    st0 = wmma_bf16(kc0, qf0, st0);
    st0 = wmma_bf16(kc1, qf1, st0);
    st1 = wmma_bf16(kc2, qf0, st1);
    st1 = wmma_bf16(kc3, qf1, st1);

    // V^T A-fragments for the current block: issue now, consume after softmax
    v16bf vf0 = load_a_frag(vh + s0,                       SEQ, lane);
    v16bf vf1 = load_a_frag(vh + (size_t)16 * SEQ + s0,    SEQ, lane);
    v16bf vf2 = load_a_frag(vh + (size_t)32 * SEQ + s0,    SEQ, lane);
    v16bf vf3 = load_a_frag(vh + (size_t)48 * SEQ + s0,    SEQ, lane);

    // Causal mask (only blocks straddling the diagonal)
    if (s0 + 31 > q0) {
      const int sb = s0 + sslot;
#pragma unroll
      for (int r = 0; r < 8; ++r) {
        if (sb + r      > qg) st0[r] = -3.0e38f;
        if (sb + 16 + r > qg) st1[r] = -3.0e38f;
      }
    }

    // Per-query chunk max: 16 local slots + partner half-wave
    float cmax = st0[0];
#pragma unroll
    for (int r = 1; r < 8; ++r) cmax = fmaxf(cmax, st0[r]);
#pragma unroll
    for (int r = 0; r < 8; ++r) cmax = fmaxf(cmax, st1[r]);
    cmax = fmaxf(cmax, __shfl_xor(cmax, 16, 32));

    const float m_new = fmaxf(m_run, cmax);
    const float scale = __expf(m_run - m_new);   // 0 on first iteration
    m_run = m_new;

    float p0[8], p1[8], s_own = 0.f;
#pragma unroll
    for (int r = 0; r < 8; ++r) {
      p0[r] = __expf(st0[r] - m_new);
      p1[r] = __expf(st1[r] - m_new);
      s_own += p0[r] + p1[r];
    }
    l_run = l_run * scale + (s_own + __shfl_xor(s_own, 16, 32));

#pragma unroll
    for (int t = 0; t < 4; ++t)
#pragma unroll
      for (int r = 0; r < 8; ++r) o[t][r] *= scale;

    // Assemble P^T B-fragment (K dim = 32 keys, N = 16 queries):
    //   lanes <16 : b[j] = own p0[j],          b[8+j] = p0[j] from lane+16
    //   lanes >=16: b[j] = p1[j] from lane-16, b[8+j] = own p1[j]
    v16bf pb;
#pragma unroll
    for (int r = 0; r < 8; ++r) {
      const float x0 = __shfl_xor(p0[r], 16, 32);
      const float x1 = __shfl_xor(p1[r], 16, 32);
      pb[r]     = f2bf(lo16 ? p0[r] : x1);
      pb[8 + r] = f2bf(lo16 ? x0 : p1[r]);
    }

    // O^T += V^T @ P^T
    o[0] = wmma_bf16(vf0, pb, o[0]);
    o[1] = wmma_bf16(vf1, pb, o[1]);
    o[2] = wmma_bf16(vf2, pb, o[2]);
    o[3] = wmma_bf16(vf3, pb, o[3]);

    kc0 = kn0; kc1 = kn1; kc2 = kn2; kc3 = kn3;
  }

  // Normalize and write attn output [B, L, D_MODEL] bf16 (col = h*64 + dim)
  const float inv = 1.0f / l_run;
  const size_t obase = ((size_t)b * SEQ + (size_t)qg) * D_MODEL + h * HEAD_DIM;
#pragma unroll
  for (int t = 0; t < 4; ++t)
#pragma unroll
    for (int r = 0; r < 8; ++r) {
      const int d = 16 * t + r + sslot;
      attnb[obase + d] = f2bf(o[t][r] * inv);
    }
}

// Output projection: out = attn @ Wo^T, fp32 result
__global__ void __launch_bounds__(128) gemm_o_kernel(const bf16_t* __restrict__ A,
                                                     const bf16_t* __restrict__ W,
                                                     float* __restrict__ out) {
  const int lane = threadIdx.x & 31;
  const int wave = blockIdx.x * (blockDim.x >> 5) + (threadIdx.x >> 5);
  const int ntn = D_MODEL / 64;
  const int m0 = (wave / ntn) * 32;
  const int n0 = (wave % ntn) * 64;
  v8f acc[8];
  gemm_32x64(A, W, D_MODEL, m0, n0, lane, acc);
  const int cb = n0 + (lane & 15);
#pragma unroll
  for (int mt = 0; mt < 2; ++mt) {
    const int rb = m0 + 16 * mt + 8 * (lane >> 4);
#pragma unroll
    for (int t = 0; t < 4; ++t) {
      const int c = cb + 16 * t;
#pragma unroll
      for (int r = 0; r < 8; ++r)
        out[(size_t)(rb + r) * D_MODEL + c] = acc[4 * mt + t][r];
    }
  }
}

// ---------------------------------------------------------------------------
extern "C" void kernel_launch(void* const* d_in, const int* in_sizes, int n_in,
                              void* d_out, int out_size, void* d_ws, size_t ws_size,
                              hipStream_t stream) {
  (void)in_sizes; (void)n_in; (void)out_size; (void)ws_size;
  const float* x  = (const float*)d_in[0];
  const float* Wq = (const float*)d_in[1];
  const float* Wk = (const float*)d_in[2];
  const float* Wv = (const float*)d_in[3];
  const float* Wo = (const float*)d_in[4];
  float* out = (float*)d_out;

  bf16_t* p   = (bf16_t*)d_ws;
  bf16_t* xb  = p; p += (size_t)MTOT * D_MODEL;
  bf16_t* wqb = p; p += (size_t)D_MODEL * D_MODEL;
  bf16_t* wkb = p; p += (size_t)KVD * D_MODEL;
  bf16_t* wvb = p; p += (size_t)KVD * D_MODEL;
  bf16_t* wob = p; p += (size_t)D_MODEL * D_MODEL;
  bf16_t* Qb  = p; p += (size_t)MTOT * D_MODEL;     // [B,H,L,64]
  bf16_t* Kb  = p; p += (size_t)MTOT * KVD;         // [B,Hkv,L,64]
  bf16_t* Vtb = p; p += (size_t)MTOT * KVD;         // [B,Hkv,64,L]
  bf16_t* atb = p; p += (size_t)MTOT * D_MODEL;     // [B,L,D] bf16

  cvt_kernel<<<2048, 256, 0, stream>>>(x,  xb,  MTOT * D_MODEL);
  cvt_kernel<<<1024, 256, 0, stream>>>(Wq, wqb, D_MODEL * D_MODEL);
  cvt_kernel<<<512,  256, 0, stream>>>(Wk, wkb, KVD * D_MODEL);
  cvt_kernel<<<512,  256, 0, stream>>>(Wv, wvb, KVD * D_MODEL);
  cvt_kernel<<<1024, 256, 0, stream>>>(Wo, wob, D_MODEL * D_MODEL);

  const int gq = (MTOT / 32) * (D_MODEL / 64) / 4;  // 1024 blocks, 4 waves each
  const int gk = (MTOT / 32) * (KVD / 64) / 4;      // 256 blocks
  gemm_q_kernel<<<gq, 128, 0, stream>>>(xb, wqb, Qb);
  gemm_k_kernel<<<gk, 128, 0, stream>>>(xb, wkb, Kb);
  gemm_v_kernel<<<gk, 128, 0, stream>>>(xb, wvb, Vtb);

  const int ga = (BATCH * NUM_HEADS * (SEQ / 16)) / 4;  // 2048 blocks
  attn_kernel<<<ga, 128, 0, stream>>>(Qb, Kb, Vtb, atb);

  gemm_o_kernel<<<gq, 128, 0, stream>>>(atb, wob, out);
}
